// TreeLSTM_65824668779099
// MI455X (gfx1250) — compile-verified, matrix operand order checked by reading, not run on hardware
//
#include <hip/hip_runtime.h>
#include <math.h>

// ---------------------------------------------------------------------------
// TreeLSTM on MI455X (gfx1250): bf16 WMMA GEMMs (v_wmma_f32_16x16x32_bf16)
// for all matmuls, fp32 accumulate. Big input GEMMs hoisted out of the
// 16-level sequential recurrence.
// ---------------------------------------------------------------------------

typedef __bf16 bf16_t;
typedef __attribute__((ext_vector_type(16))) __bf16 v16bf;
typedef __attribute__((ext_vector_type(8)))  __bf16 v8bf;
typedef __attribute__((ext_vector_type(8)))  float  v8f;

#define NLEV   16
#define MLVL   1024
#define NNODE  16384
#define KCH    4
#define DIN    1024
#define HID    256
#define IOU    768   /* 3*HID */

// ----------------------------- fragment loader -----------------------------
// Row-major [rows, ld] bf16 source. Produces the 16-bit A-operand layout
// (16x32 tile): lane L holds row (row0 + L%16); K-chunks at
// k0 + 8*(L/16) .. +7  (elements 0..7)  and  k0 + 16 + 8*(L/16) .. +7
// (elements 8..15). The same loader applied to the row-major weight [N,K]
// (i.e. B^T) yields the B operand for D = A x B.
__device__ __forceinline__ v16bf load_frag(const bf16_t* __restrict__ base,
                                           int ld, int row0, int k0) {
    int lane = threadIdx.x & 31;
    const bf16_t* p = base + (size_t)(row0 + (lane & 15)) * ld
                           + k0 + ((lane >> 4) << 3);
    v8bf lo = *(const v8bf*)(p);
    v8bf hi = *(const v8bf*)(p + 16);
    v16bf r;
#pragma unroll
    for (int i = 0; i < 8; ++i) { r[i] = lo[i]; r[i + 8] = hi[i]; }
    return r;
}

__device__ __forceinline__ float sigmoidf_(float x) {
    return 1.0f / (1.0f + __expf(-x));
}

// --------------------------- f32 -> bf16 convert ---------------------------
__global__ __launch_bounds__(256) void cvt_f32_bf16(const float* __restrict__ in,
                                                    bf16_t* __restrict__ out,
                                                    int n) {
    int i = blockIdx.x * 256 + threadIdx.x;
    if (i < n) out[i] = (bf16_t)in[i];
}

// ------------------------------ WMMA GEMM ----------------------------------
// out[M, ldout(=N)] = A[M,K](bf16) @ W[N,K]^T(bf16) + bias[N]
//                     (+ extra[(extra_rowbase + row>>shift)*extra_ld + col])
// Wave tile 32x32 (2x2 WMMA frags), 8 waves/block -> 64x128 block tile.
// grid = (N/128, M/64), block = 256.
__global__ __launch_bounds__(256) void gemm_bf16(
    const bf16_t* __restrict__ A,
    const bf16_t* __restrict__ W,
    float* __restrict__ out, int K, int ldout,
    const float* __restrict__ bias,
    const float* __restrict__ extra, int extra_ld,
    int extra_rowshift, int extra_rowbase)
{
    int lane = threadIdx.x & 31;
    int wave = threadIdx.x >> 5;
    int wm = wave >> 2;          // 0..1
    int wn = wave & 3;           // 0..3
    int m0 = blockIdx.y * 64 + wm * 32;
    int n0 = blockIdx.x * 128 + wn * 32;
    int colq = lane & 15;
    int rowq = (lane >> 4) << 3;

    v8f acc[2][2];
#pragma unroll
    for (int i = 0; i < 2; ++i)
#pragma unroll
        for (int j = 0; j < 2; ++j) {
            int col = n0 + j * 16 + colq;
            float bv = bias[col];
#pragma unroll
            for (int r = 0; r < 8; ++r) {
                float v = bv;
                if (extra) {
                    int row = m0 + i * 16 + rowq + r;
                    v += extra[(size_t)(extra_rowbase + (row >> extra_rowshift))
                               * extra_ld + col];
                }
                acc[i][j][r] = v;
            }
        }

    for (int k = 0; k < K; k += 32) {
        v16bf a0 = load_frag(A, K, m0,      k);
        v16bf a1 = load_frag(A, K, m0 + 16, k);
        v16bf b0 = load_frag(W, K, n0,      k);
        v16bf b1 = load_frag(W, K, n0 + 16, k);
        acc[0][0] = __builtin_amdgcn_wmma_f32_16x16x32_bf16(
            false, a0, false, b0, (short)0, acc[0][0], false, false);
        acc[0][1] = __builtin_amdgcn_wmma_f32_16x16x32_bf16(
            false, a0, false, b1, (short)0, acc[0][1], false, false);
        acc[1][0] = __builtin_amdgcn_wmma_f32_16x16x32_bf16(
            false, a1, false, b0, (short)0, acc[1][0], false, false);
        acc[1][1] = __builtin_amdgcn_wmma_f32_16x16x32_bf16(
            false, a1, false, b1, (short)0, acc[1][1], false, false);
    }

#pragma unroll
    for (int i = 0; i < 2; ++i)
#pragma unroll
        for (int j = 0; j < 2; ++j)
#pragma unroll
            for (int r = 0; r < 8; ++r)
                out[(size_t)(m0 + i * 16 + rowq + r) * ldout
                    + n0 + j * 16 + colq] = acc[i][j][r];
}

// ----------------------- per-level child gather ----------------------------
// One block per node in the level; 256 threads = H lanes.
__global__ __launch_bounds__(256) void gather_children(
    const int* __restrict__ children, const float* __restrict__ h,
    bf16_t* __restrict__ hsum_b, bf16_t* __restrict__ hch_b, int level)
{
    int m = blockIdx.x, t = threadIdx.x;
    int node = level * MLVL + m;
    float s = 0.f;
#pragma unroll
    for (int k = 0; k < KCH; ++k) {
        int ch = children[node * KCH + k];
        float v = 0.f;
        if (ch >= 0) v = h[(size_t)ch * HID + t];
        s += v;
        hch_b[(size_t)(m * KCH + k) * HID + t] = (bf16_t)v;
    }
    hsum_b[(size_t)m * HID + t] = (bf16_t)s;
}

// --------------------------- gates / state update --------------------------
__global__ __launch_bounds__(256) void tree_lstm_pointwise(
    const float* __restrict__ iou,   // [MLVL, 768] pre-activation (biases fused)
    const float* __restrict__ fh,    // [MLVL*KCH, HID] pre-activation
    const int* __restrict__ children,
    float* __restrict__ c, float* __restrict__ h, int level)
{
    int m = blockIdx.x, t = threadIdx.x;
    int node = level * MLVL + m;
    float ig = sigmoidf_(iou[(size_t)m * IOU + t]);
    float og = sigmoidf_(iou[(size_t)m * IOU + HID + t]);
    float ug = tanhf(iou[(size_t)m * IOU + 2 * HID + t]);
    float csum = 0.f;
#pragma unroll
    for (int k = 0; k < KCH; ++k) {
        int ch = children[node * KCH + k];
        if (ch >= 0) {
            float f = sigmoidf_(fh[(size_t)(m * KCH + k) * HID + t]);
            csum += f * c[(size_t)ch * HID + t];
        }
    }
    float cn = ig * ug + csum;
    c[(size_t)node * HID + t] = cn;
    h[(size_t)node * HID + t] = og * tanhf(cn);
}

// ------------------------------- launcher ----------------------------------
extern "C" void kernel_launch(void* const* d_in, const int* in_sizes, int n_in,
                              void* d_out, int out_size, void* d_ws, size_t ws_size,
                              hipStream_t stream) {
    const float* x        = (const float*)d_in[0];
    const int*   children = (const int*)  d_in[1];
    const float* Wioux    = (const float*)d_in[2];
    const float* bioux    = (const float*)d_in[3];
    const float* Wiouh    = (const float*)d_in[4];
    const float* biouh    = (const float*)d_in[5];
    const float* Wfx      = (const float*)d_in[6];
    const float* bfx      = (const float*)d_in[7];
    const float* Wfh      = (const float*)d_in[8];
    const float* bfh      = (const float*)d_in[9];
    float* h_out = (float*)d_out;
    (void)in_sizes; (void)n_in; (void)out_size; (void)ws_size;

    char* ws = (char*)d_ws;
    size_t off = 0;
    auto take = [&](size_t bytes) -> char* {
        char* p = ws + off;
        off = (off + bytes + 255) & ~(size_t)255;
        return p;
    };

    bf16_t* xb       = (bf16_t*)take((size_t)NNODE * DIN * 2);
    bf16_t* Wioux_b  = (bf16_t*)take((size_t)IOU * DIN * 2);
    bf16_t* Wfx_b    = (bf16_t*)take((size_t)HID * DIN * 2);
    bf16_t* Wiouh_b  = (bf16_t*)take((size_t)IOU * HID * 2);
    bf16_t* Wfh_b    = (bf16_t*)take((size_t)HID * HID * 2);
    float*  iou_x    = (float*) take((size_t)NNODE * IOU * 4);
    float*  f_x      = (float*) take((size_t)NNODE * HID * 4);
    float*  cbuf     = (float*) take((size_t)NNODE * HID * 4);
    bf16_t* hsum_b   = (bf16_t*)take((size_t)MLVL * HID * 2);
    bf16_t* hch_b    = (bf16_t*)take((size_t)MLVL * KCH * HID * 2);
    float*  iou_lvl  = (float*) take((size_t)MLVL * IOU * 4);
    float*  fh_lvl   = (float*) take((size_t)MLVL * KCH * HID * 4);

    auto cvt = [&](const float* src, bf16_t* dst, int n) {
        cvt_f32_bf16<<<(n + 255) / 256, 256, 0, stream>>>(src, dst, n);
    };
    cvt(x,     xb,      NNODE * DIN);
    cvt(Wioux, Wioux_b, IOU * DIN);
    cvt(Wfx,   Wfx_b,   HID * DIN);
    cvt(Wiouh, Wiouh_b, IOU * HID);
    cvt(Wfh,   Wfh_b,   HID * HID);

    // iou_x = x @ Wioux^T + bioux   [16384, 768], K=1024
    gemm_bf16<<<dim3(IOU / 128, NNODE / 64), 256, 0, stream>>>(
        xb, Wioux_b, iou_x, DIN, IOU, bioux, nullptr, 0, 0, 0);
    // f_x = x @ Wfx^T + bfx         [16384, 256], K=1024
    gemm_bf16<<<dim3(HID / 128, NNODE / 64), 256, 0, stream>>>(
        xb, Wfx_b, f_x, DIN, HID, bfx, nullptr, 0, 0, 0);

    for (int l = 0; l < NLEV; ++l) {
        gather_children<<<MLVL, 256, 0, stream>>>(children, h_out, hsum_b, hch_b, l);

        // iou_lvl = hsum @ Wiouh^T + biouh + iou_x[level]   [1024, 768], K=256
        gemm_bf16<<<dim3(IOU / 128, MLVL / 64), 256, 0, stream>>>(
            hsum_b, Wiouh_b, iou_lvl, HID, IOU, biouh,
            iou_x, IOU, /*shift=*/0, /*rowbase=*/l * MLVL);

        // fh_lvl = h_ch @ Wfh^T + bfh + f_x[level, broadcast per child]
        // [4096, 256], K=256; row>>2 maps child-row -> node-in-level
        gemm_bf16<<<dim3(HID / 128, (MLVL * KCH) / 64), 256, 0, stream>>>(
            hch_b, Wfh_b, fh_lvl, HID, HID, bfh,
            f_x, HID, /*shift=*/2, /*rowbase=*/l * MLVL);

        tree_lstm_pointwise<<<MLVL, 256, 0, stream>>>(
            iou_lvl, fh_lvl, children, cbuf, h_out, l);
    }
}
